// Network_68753836474807
// MI455X (gfx1250) — matrix operand order repaired and checked by reading, not compiled, not run on hardware
//
#include <hip/hip_runtime.h>
#include <stdint.h>

// ---------------------------------------------------------------------------
// Faster-RCNN proposal head for MI455X (gfx1250).
// 3x3 conv -> implicit GEMM on v_wmma_f32_16x16x32_bf16 (wave32 WMMA),
// with async global->LDS staging of the shared B tile (ASYNCcnt pipeline).
// M=2394(->2432), N=512, K=4608.
// ---------------------------------------------------------------------------

#define FH    38
#define FW    63
#define CIN   512
#define COUT  512
#define MREAL (FH*FW)          // 2394
#define MPAD  2432             // 152 tiles of 16 -> 38 groups of 4
#define KDIM  (CIN*9)          // 4608
#define NA    9
#define NPROP (MREAL*NA)       // 21546
#define PRE_NMS  6000
#define POST_NMS 300
#define NMS_TH   0.7f
#define CROP  14
#define POOL  7

typedef __attribute__((ext_vector_type(16))) __bf16 v16bf;
typedef __attribute__((ext_vector_type(8)))  float  v8f;
typedef __attribute__((ext_vector_type(4)))  unsigned int v4u;

union V16 { v4u q[2]; v16bf v; };

// --------------------------- im2col (f32 -> bf16) --------------------------
__global__ void im2col_k(const float* __restrict__ x, __bf16* __restrict__ A, int total) {
    int i = blockIdx.x * blockDim.x + threadIdx.x;
    if (i >= total) return;
    int m = i / KDIM, k = i - m * KDIM;
    int c = k / 9, r = k - c * 9;
    int ky = r / 3, kx = r - ky * 3;
    int y = m / FW + ky - 1;
    int xx = m % FW + kx - 1;
    float v = 0.0f;
    if (m < MREAL && (unsigned)y < FH && (unsigned)xx < FW)
        v = x[(c * FH + y) * FW + xx];
    A[i] = (__bf16)v;
}

// ------------------------- weight convert (f32->bf16) ----------------------
__global__ void cvt_k(const float* __restrict__ w, __bf16* __restrict__ B, int n) {
    int i = blockIdx.x * blockDim.x + threadIdx.x;
    if (i < n) B[i] = (__bf16)w[i];
}

// -------------------- WMMA GEMM: rpn = relu(A*B^T + bias) ------------------
// A: [MPAD][KDIM] bf16 row-major (im2col).  B: [COUT][KDIM] bf16 row-major.
// Block: 4 waves, one ngroup (64 cols) shared, 4 consecutive mtiles.
// B tile (64 x 32 bf16 = 4KB) staged in double-buffered LDS via
// global_load_async_to_lds_b128 (ASYNCcnt), one k-step ahead.
__global__ __launch_bounds__(128) void rpn_gemm_k(const __bf16* __restrict__ A,
                                                  const __bf16* __restrict__ B,
                                                  const float*  __restrict__ bias,
                                                  float* __restrict__ out) {
    __shared__ __align__(16) unsigned char ldsB[2][4096];   // [buf][64 rows * 64B]

    const int tid   = threadIdx.x;
    const int lane  = tid & 31;
    const int wave  = tid >> 5;
    const int ngrp  = blockIdx.x & 7;                 // 0..7 -> 64 cols each
    const int mtile = (blockIdx.x >> 3) * 4 + wave;   // 0..151
    const int l15   = lane & 15;
    const int hi    = lane >> 4;                      // 0: lanes 0-15, 1: lanes 16-31

    // Cooperative async-copy assignment: 4KB = 256 x 16B chunks, 2 per thread.
    const int     row0 = tid >> 2,          kc0 = tid & 3;   // rows 0..31
    const int     row1 = (tid + 128) >> 2,  kc1 = tid & 3;   // rows 32..63
    const unsigned ldsOff0 = (unsigned)(row0 * 64 + kc0 * 16);
    const unsigned ldsOff1 = (unsigned)(row1 * 64 + kc1 * 16);
    const unsigned gOff0   = (unsigned)(row0 * KDIM * 2 + kc0 * 16);
    const unsigned gOff1   = (unsigned)(row1 * KDIM * 2 + kc1 * 16);
    const unsigned long long sB =
        (unsigned long long)(uintptr_t)(B + (size_t)ngrp * 64 * KDIM);

    // A fragment (16x32 bf16): lanes 0-15 row M=l15 K{0..15},
    // lanes 16-31 same row K{16..31}.
    const __bf16* aRow = A + (size_t)(mtile * 16 + l15) * KDIM + hi * 16;

    v8f acc0 = {}, acc1 = {}, acc2 = {}, acc3 = {};

#define ISSUE_COPY(ksNext, bufNext)                                              \
    do {                                                                         \
        unsigned base = (unsigned)(uintptr_t)&ldsB[(bufNext)][0];                \
        unsigned ksb  = (unsigned)((ksNext) * 2);                                \
        unsigned l0 = base + ldsOff0, l1 = base + ldsOff1;                       \
        unsigned o0 = gOff0 + ksb,    o1 = gOff1 + ksb;                          \
        asm volatile("global_load_async_to_lds_b128 %0, %1, %2"                  \
                     :: "v"(l0), "v"(o0), "s"(sB) : "memory");                   \
        asm volatile("global_load_async_to_lds_b128 %0, %1, %2"                  \
                     :: "v"(l1), "v"(o1), "s"(sB) : "memory");                   \
    } while (0)

    ISSUE_COPY(0, 0);

    // Register double-buffer for the A fragment.
    v4u a0 = *(const v4u*)(aRow + 0);
    v4u a1 = *(const v4u*)(aRow + 8);

    int buf = 0;
    for (int ks = 0; ks < KDIM; ks += 32, buf ^= 1) {
        // Our async chunks for this buffer are done; sync so everyone's are.
        asm volatile("s_wait_asynccnt 0" ::: "memory");
        __syncthreads();

        // Kick the copy for the next k-step into the other buffer.
        if (ks + 32 < KDIM) ISSUE_COPY(ks + 32, buf ^ 1);

        // Preload next A fragment (overlaps with this step's WMMAs).
        v4u na0 = a0, na1 = a1;
        if (ks + 32 < KDIM) {
            na0 = *(const v4u*)(aRow + ks + 32);
            na1 = *(const v4u*)(aRow + ks + 40);
        }

        V16 ua; ua.q[0] = a0; ua.q[1] = a1;
        const unsigned char* bbuf = &ldsB[buf][0];

        // B fragment (32x16): row n, lanes 0-15 bytes [0,32) = K 0..15,
        // lanes 16-31 bytes [32,64) = K 16..31.
        V16 ub;
        const v4u* p;
        p = (const v4u*)(bbuf + (l15 +  0) * 64 + hi * 32);
        ub.q[0] = p[0]; ub.q[1] = p[1];
        acc0 = __builtin_amdgcn_wmma_f32_16x16x32_bf16(false, ua.v, false, ub.v,
                                                       (short)0, acc0, false, false);
        p = (const v4u*)(bbuf + (l15 + 16) * 64 + hi * 32);
        ub.q[0] = p[0]; ub.q[1] = p[1];
        acc1 = __builtin_amdgcn_wmma_f32_16x16x32_bf16(false, ua.v, false, ub.v,
                                                       (short)0, acc1, false, false);
        p = (const v4u*)(bbuf + (l15 + 32) * 64 + hi * 32);
        ub.q[0] = p[0]; ub.q[1] = p[1];
        acc2 = __builtin_amdgcn_wmma_f32_16x16x32_bf16(false, ua.v, false, ub.v,
                                                       (short)0, acc2, false, false);
        p = (const v4u*)(bbuf + (l15 + 48) * 64 + hi * 32);
        ub.q[0] = p[0]; ub.q[1] = p[1];
        acc3 = __builtin_amdgcn_wmma_f32_16x16x32_bf16(false, ua.v, false, ub.v,
                                                       (short)0, acc3, false, false);

        a0 = na0; a1 = na1;
    }
#undef ISSUE_COPY

    // C/D layout: VGPR r -> M = r + 8*hi, N = l15.  Fused bias + ReLU.
    const int ncol = ngrp * 64 + l15;
    const float bv0 = bias[ncol], bv1 = bias[ncol + 16];
    const float bv2 = bias[ncol + 32], bv3 = bias[ncol + 48];
#pragma unroll
    for (int r = 0; r < 8; ++r) {
        int m = mtile * 16 + r + hi * 8;
        if (m >= MREAL) continue;
        float* o = out + (size_t)m * COUT + ncol;
        float v;
        v = acc0[r] + bv0; o[0]  = v > 0.f ? v : 0.f;
        v = acc1[r] + bv1; o[16] = v > 0.f ? v : 0.f;
        v = acc2[r] + bv2; o[32] = v > 0.f ? v : 0.f;
        v = acc3[r] + bv3; o[48] = v > 0.f ? v : 0.f;
    }
}

// --------------------------- 1x1 convs (cls + bbox) ------------------------
__global__ void conv1x1_k(const float* __restrict__ rpn,
                          const float* __restrict__ wc, const float* __restrict__ bc,
                          const float* __restrict__ wb, const float* __restrict__ bb,
                          float* __restrict__ cls, float* __restrict__ bbx) {
    int i = blockIdx.x * blockDim.x + threadIdx.x;
    if (i >= MREAL * 54) return;
    int m = i / 54, ch = i - m * 54;
    const float* row = rpn + (size_t)m * COUT;
    const float* w;
    float s;
    if (ch < 18) { w = wc + ch * COUT;        s = bc[ch]; }
    else         { w = wb + (ch - 18) * COUT; s = bb[ch - 18]; }
    for (int c = 0; c < COUT; ++c) s += row[c] * w[c];
    if (ch < 18) cls[m * 18 + ch]        = s;
    else         bbx[m * 36 + (ch - 18)] = s;
}

// ------------- scores (2-way softmax == sigmoid) + bbox decode + clip ------
__global__ void props_k(const float* __restrict__ cls, const float* __restrict__ bbx,
                        const float* __restrict__ anchors,
                        float* __restrict__ scores, float* __restrict__ props) {
    int i = blockIdx.x * blockDim.x + threadIdx.x;
    if (i >= NPROP) return;
    int m = i / NA, a = i - m * NA;
    float bg = cls[m * 18 + a];
    float fg = cls[m * 18 + NA + a];
    scores[i] = 1.0f / (1.0f + expf(bg - fg));

    float dx = bbx[m * 36 + a * 4 + 0];
    float dy = bbx[m * 36 + a * 4 + 1];
    float dw = bbx[m * 36 + a * 4 + 2];
    float dh = bbx[m * 36 + a * 4 + 3];
    const float* an = anchors + (size_t)i * 4;
    float w  = an[2] - an[0] + 1.0f;
    float h  = an[3] - an[1] + 1.0f;
    float cx = an[0] + 0.5f * w;
    float cy = an[1] + 0.5f * h;
    float pcx = dx * w + cx, pcy = dy * h + cy;
    float pw  = expf(dw) * w, ph = expf(dh) * h;
    props[i * 4 + 0] = fminf(fmaxf(pcx - 0.5f * pw, 0.0f), 999.0f);
    props[i * 4 + 1] = fminf(fmaxf(pcy - 0.5f * ph, 0.0f), 599.0f);
    props[i * 4 + 2] = fminf(fmaxf(pcx + 0.5f * pw, 0.0f), 999.0f);
    props[i * 4 + 3] = fminf(fmaxf(pcy + 0.5f * ph, 0.0f), 599.0f);
}

// --------- exact top-k by rank (matches lax.top_k ordering/ties) -----------
__global__ void topk_k(const float* __restrict__ scores, const float* __restrict__ props,
                       float* __restrict__ sorted) {
    int i = blockIdx.x * blockDim.x + threadIdx.x;
    if (i >= NPROP) return;
    float s = scores[i];
    int rank = 0;
    for (int j = 0; j < NPROP; ++j) {
        float sj = scores[j];
        if (sj > s || (sj == s && j < i)) ++rank;
    }
    if (rank < PRE_NMS) {
        sorted[rank * 4 + 0] = props[i * 4 + 0];
        sorted[rank * 4 + 1] = props[i * 4 + 1];
        sorted[rank * 4 + 2] = props[i * 4 + 2];
        sorted[rank * 4 + 3] = props[i * 4 + 3];
    }
}

// ------------------- greedy NMS (sequential, 1 workgroup) ------------------
__global__ __launch_bounds__(1024) void nms_k(const float* __restrict__ boxes,
                                              float* __restrict__ rois) {
    __shared__ int keep[PRE_NMS];
    const int tid = threadIdx.x;
    const float4* b4 = (const float4*)boxes;

    float4 my[6];
    float  myarea[6];
#pragma unroll
    for (int s = 0; s < 6; ++s) {
        int j = tid + s * 1024;
        if (j < PRE_NMS) {
            my[s] = b4[j];
            myarea[s] = (my[s].z - my[s].x + 1.0f) * (my[s].w - my[s].y + 1.0f);
            keep[j] = 1;
        }
    }
    __syncthreads();

    for (int i = 0; i < PRE_NMS; ++i) {
        __syncthreads();
        if (!keep[i]) continue;                 // uniform: same LDS word for all
        float4 bi = b4[i];                      // broadcast load
        float  ai = (bi.z - bi.x + 1.0f) * (bi.w - bi.y + 1.0f);
#pragma unroll
        for (int s = 0; s < 6; ++s) {
            int j = tid + s * 1024;
            if (j > i && j < PRE_NMS) {
                float iw = fminf(bi.z, my[s].z) - fmaxf(bi.x, my[s].x) + 1.0f;
                float ih = fminf(bi.w, my[s].w) - fmaxf(bi.y, my[s].y) + 1.0f;
                iw = fmaxf(iw, 0.0f); ih = fmaxf(ih, 0.0f);
                float inter = iw * ih;
                float iou = inter / (ai + myarea[s] - inter);
                if (iou > NMS_TH) keep[j] = 0;
            }
        }
    }
    __syncthreads();

    if (tid == 0) {   // first POST_NMS kept (ascending rank), pad with box[5999]
        int cnt = 0;
        for (int i = 0; i < PRE_NMS && cnt < POST_NMS; ++i) {
            if (keep[i]) {
                rois[cnt * 4 + 0] = boxes[i * 4 + 0];
                rois[cnt * 4 + 1] = boxes[i * 4 + 1];
                rois[cnt * 4 + 2] = boxes[i * 4 + 2];
                rois[cnt * 4 + 3] = boxes[i * 4 + 3];
                ++cnt;
            }
        }
        for (; cnt < POST_NMS; ++cnt) {
            rois[cnt * 4 + 0] = boxes[(PRE_NMS - 1) * 4 + 0];
            rois[cnt * 4 + 1] = boxes[(PRE_NMS - 1) * 4 + 1];
            rois[cnt * 4 + 2] = boxes[(PRE_NMS - 1) * 4 + 2];
            rois[cnt * 4 + 3] = boxes[(PRE_NMS - 1) * 4 + 3];
        }
    }
}

// -------------- crop_and_resize (14x14 bilinear) + 2x2 max pool ------------
__global__ void crop_k(const float* __restrict__ feat, const float* __restrict__ rois,
                       float* __restrict__ out, int total) {
    int idx = blockIdx.x * blockDim.x + threadIdx.x;
    if (idx >= total) return;
    int px = idx % POOL;
    int py = (idx / POOL) % POOL;
    int c  = (idx / (POOL * POOL)) % CIN;
    int r  = idx / (POOL * POOL * CIN);

    const float* rb = rois + r * 4;
    float bx1 = (rb[0] * (1.0f / 16.0f)) / (float)(FW - 1);
    float by1 = (rb[1] * (1.0f / 16.0f)) / (float)(FH - 1);
    float bx2 = (rb[2] * (1.0f / 16.0f)) / (float)(FW - 1);
    float by2 = (rb[3] * (1.0f / 16.0f)) / (float)(FH - 1);

    const float* f = feat + (size_t)c * FH * FW;
    float mx = -3.0e38f;
#pragma unroll
    for (int dy = 0; dy < 2; ++dy) {
#pragma unroll
        for (int dx = 0; dx < 2; ++dx) {
            float ty = (float)(2 * py + dy) * (1.0f / (CROP - 1));
            float tx = (float)(2 * px + dx) * (1.0f / (CROP - 1));
            float ys = by1 * (FH - 1) + ty * (by2 - by1) * (FH - 1);
            float xs = bx1 * (FW - 1) + tx * (bx2 - bx1) * (FW - 1);
            float y0f = floorf(ys), x0f = floorf(xs);
            float wy = ys - y0f, wx = xs - x0f;
            int y0 = (int)fminf(fmaxf(y0f,        0.0f), (float)(FH - 1));
            int y1 = (int)fminf(fmaxf(y0f + 1.0f, 0.0f), (float)(FH - 1));
            int x0 = (int)fminf(fmaxf(x0f,        0.0f), (float)(FW - 1));
            int x1 = (int)fminf(fmaxf(x0f + 1.0f, 0.0f), (float)(FW - 1));
            float f00 = f[y0 * FW + x0], f01 = f[y0 * FW + x1];
            float f10 = f[y1 * FW + x0], f11 = f[y1 * FW + x1];
            float v = f00 * (1 - wy) * (1 - wx) + f01 * (1 - wy) * wx
                    + f10 * wy * (1 - wx)       + f11 * wy * wx;
            mx = fmaxf(mx, v);
        }
    }
    out[idx] = mx;
}

// ---------------------------------------------------------------------------
extern "C" void kernel_launch(void* const* d_in, const int* in_sizes, int n_in,
                              void* d_out, int out_size, void* d_ws, size_t ws_size,
                              hipStream_t stream) {
    const float* net     = (const float*)d_in[0];   // (1,512,38,63)
    const float* w_rpn   = (const float*)d_in[1];   // (512,512,3,3) == B[N][K]
    const float* b_rpn   = (const float*)d_in[2];   // (512,)
    const float* w_cls   = (const float*)d_in[3];   // (18,512,1,1)
    const float* b_cls   = (const float*)d_in[4];   // (18,)
    const float* w_bbox  = (const float*)d_in[5];   // (36,512,1,1)
    const float* b_bbox  = (const float*)d_in[6];   // (36,)
    const float* anchors = (const float*)d_in[7];   // (21546,4)
    float* out = (float*)d_out;                     // (300,512,7,7)

    char* ws = (char*)d_ws;
    auto alloc = [&](size_t bytes) -> char* {
        char* p = ws;
        ws += (bytes + 255) & ~(size_t)255;
        return p;
    };
    __bf16* Amat  = (__bf16*)alloc((size_t)MPAD * KDIM * 2);
    __bf16* Bmat  = (__bf16*)alloc((size_t)COUT * KDIM * 2);
    float*  rpn   = (float*) alloc((size_t)MREAL * COUT * 4);
    float*  clsO  = (float*) alloc((size_t)MREAL * 18 * 4);
    float*  bbO   = (float*) alloc((size_t)MREAL * 36 * 4);
    float*  sc    = (float*) alloc((size_t)NPROP * 4);
    float*  pr    = (float*) alloc((size_t)NPROP * 4 * 4);
    float*  srt   = (float*) alloc((size_t)PRE_NMS * 4 * 4);
    float*  rois  = (float*) alloc((size_t)POST_NMS * 4 * 4);

    const int tA = MPAD * KDIM;
    im2col_k<<<(tA + 255) / 256, 256, 0, stream>>>(net, Amat, tA);
    const int tB = COUT * KDIM;
    cvt_k<<<(tB + 255) / 256, 256, 0, stream>>>(w_rpn, Bmat, tB);

    rpn_gemm_k<<<(MPAD / 16 / 4) * 8, 128, 0, stream>>>(Amat, Bmat, b_rpn, rpn);

    conv1x1_k<<<(MREAL * 54 + 255) / 256, 256, 0, stream>>>(rpn, w_cls, b_cls,
                                                            w_bbox, b_bbox, clsO, bbO);
    props_k<<<(NPROP + 255) / 256, 256, 0, stream>>>(clsO, bbO, anchors, sc, pr);
    topk_k<<<(NPROP + 255) / 256, 256, 0, stream>>>(sc, pr, srt);
    nms_k<<<1, 1024, 0, stream>>>(srt, rois);

    const int tO = POST_NMS * CIN * POOL * POOL;
    crop_k<<<(tO + 255) / 256, 256, 0, stream>>>(net, rois, out, tO);
}
